// model_12043088298405
// MI455X (gfx1250) — matrix operand order repaired and checked by reading, not compiled, not run on hardware
//
#include <hip/hip_runtime.h>
#include <hip/hip_bf16.h>

#define DEV __device__ __forceinline__

#define B_   8
#define T_   256
#define NE_  64
#define H_   512
#define H2_  1024
#define G4_  2048
#define NH_  4
#define HD_  128
#define V_   32000
#define OUTC (V_ + NE_)   // 32064

typedef __attribute__((ext_vector_type(16))) __bf16 v16bf;
typedef __attribute__((ext_vector_type(8)))  float  v8f;
typedef unsigned int u32x4 __attribute__((ext_vector_type(4)));
typedef int          i32x4 __attribute__((ext_vector_type(4)));
typedef int          i32x8 __attribute__((ext_vector_type(8)));

union BfVec { v16bf v; unsigned u32[8]; unsigned short u16[16]; };

DEV unsigned short f32_to_bf16(float f) {
  union { float f; unsigned u; } x; x.f = f;
  unsigned r = x.u + 0x7FFFu + ((x.u >> 16) & 1u);   // RNE
  return (unsigned short)(r >> 16);
}

DEV float sigmoidf(float x) { return 1.f / (1.f + __expf(-x)); }

// Unguarded A fragment: 16x32 bf16, row-major (lda elements), rows 0..15 valid.
// Layout: lanes 0-15 hold row M=lane, K=0..15; lanes 16-31 K=16..31.
// Loads land as two 16B chunks per lane -> b128-mergeable.
DEV v16bf load_a_frag(const unsigned short* A, int lda, int k0) {
  int lane = threadIdx.x & 31;
  const unsigned short* p =
      A + (size_t)(lane & 15) * lda + k0 + ((lane >> 4) ? 16 : 0);
  const unsigned* p0 = reinterpret_cast<const unsigned*>(p);
  const unsigned* p1 = reinterpret_cast<const unsigned*>(p + 8);
  BfVec u;
#pragma unroll
  for (int vv = 0; vv < 4; ++vv) { u.u32[vv] = p0[vv]; u.u32[4 + vv] = p1[vv]; }
  return u.v;
}

// B fragment for D = A @ W^T: B[k][n] = W[n][k], W row-major NxK (ldb=K).
// lanes 0-15: col N=lane, K=0..15; lanes 16-31: K=16..31.
DEV v16bf load_b_frag(const unsigned short* W, int ldb, int n0, int k0) {
  int lane = threadIdx.x & 31;
  const unsigned* p = reinterpret_cast<const unsigned*>(
      W + (size_t)(n0 + (lane & 15)) * ldb + k0 + ((lane >> 4) ? 16 : 0));
  BfVec u;
#pragma unroll
  for (int vv = 0; vv < 8; ++vv) u.u32[vv] = p[vv];
  return u.v;
}

DEV v8f wmma_bf16(v16bf a, v16bf b, v8f c) {
  return __builtin_amdgcn_wmma_f32_16x16x32_bf16(false, a, false, b, (short)0, c, false, false);
}

// -------- Tensor Data Mover: stage a 2D bf16 tile (rows x row_elems) into LDS.
// D# per CDNA5 ISA ch.8: group0 = {count|flags, lds_addr, global_addr, type=2},
// group1 = {data_size=2B, tensor_dim0/1, tile_dim0/1, tensor_dim0_stride}.
// This toolchain exposes the 6-arg builtin: (g0, g1, g2, g3, g4, cpol).
DEV void tdm_load_2d_bf16(unsigned lds_byte_addr, const unsigned short* gptr,
                          int row_elems, int rows, int stride_elems,
                          int tensor_rows) {
  unsigned long long ga = (unsigned long long)(uintptr_t)gptr;
  union { u32x4 v; unsigned u[4]; } g0;
  g0.u[0] = 1u;                                       // count=1, user descriptor
  g0.u[1] = lds_byte_addr;                            // lds_addr
  g0.u[2] = (unsigned)(ga & 0xFFFFFFFFu);             // global_addr[31:0]
  g0.u[3] = (unsigned)((ga >> 32) & 0x1FFFFFFu)       // global_addr[56:32]
            | (2u << 30);                             // type=2 ("image")
  union { i32x8 v; unsigned u[8]; } g1;
  unsigned td0 = (unsigned)row_elems;                 // tensor_dim0
  unsigned td1 = (unsigned)tensor_rows;               // tensor_dim1 (OOB bound)
  unsigned long long s0 = (unsigned long long)stride_elems;
  g1.u[0] = (1u << 16);                               // data_size=1 -> 2 bytes
  g1.u[1] = (td0 & 0xFFFFu) << 16;                    // tensor_dim0[15:0]
  g1.u[2] = (td0 >> 16) | ((td1 & 0xFFFFu) << 16);    // dim0 hi | dim1 lo
  g1.u[3] = (td1 >> 16) | ((unsigned)(row_elems & 0xFFFF) << 16); // tile_dim0
  g1.u[4] = (unsigned)(rows & 0xFFFF);                // tile_dim1 (tile_dim2=0)
  g1.u[5] = (unsigned)(s0 & 0xFFFFFFFFu);             // tensor_dim0_stride lo
  g1.u[6] = (unsigned)((s0 >> 32) & 0xFFFFu);         // stride hi | dim1_stride=0
  g1.u[7] = 0;
  i32x4 z4 = {0, 0, 0, 0};
  i32x8 z8 = {0, 0, 0, 0, 0, 0, 0, 0};
  __builtin_amdgcn_tensor_load_to_lds(g0.v, g1.v, z4, z4, z8, 0);
}

// ---------------------------------------------------------------- converts
__global__ void to_bf16_kernel(const float* __restrict__ a,
                               unsigned short* __restrict__ o, long n) {
  long i = (long)blockIdx.x * blockDim.x + threadIdx.x;
  long stride = (long)gridDim.x * blockDim.x;
  for (; i < n; i += stride) o[i] = f32_to_bf16(a[i]);
}

__global__ void gather_kernel(const int* __restrict__ outp,
                              const float* __restrict__ emb,
                              unsigned short* __restrict__ e_bf) {
  int row = blockIdx.x;                       // b*T + t
  int tok = outp[row];
  const float* src = emb + (size_t)tok * H_;
  unsigned short* dst = e_bf + (size_t)row * H_;
  for (int h = threadIdx.x; h < H_; h += blockDim.x) dst[h] = f32_to_bf16(src[h]);
}

// ------------------------------------------------- generic C = A @ W^T + b
// One wave per block computes NTILES consecutive 16x16 tiles of one M-row-tile.
// The 16xK A slab is staged into LDS once via the Tensor Data Mover and reused
// across all NTILES WMMAs per k-step. M must be padded by tensor OOB (zeros).
// grid = (N/(16*NTILES), M/16, batch). K <= 1024.
template <int NTILES>
__global__ void __launch_bounds__(32)
wmma_gemm_bt(const unsigned short* __restrict__ A,
             const unsigned short* __restrict__ W,
             const float* __restrict__ bias,
             float* __restrict__ C,
             int M, int N, int K, int ldc,
             long strideA, long strideC) {
  __shared__ unsigned short lds_a[16 * H2_];          // 32 KB staged A tile
  const unsigned short* Ab = A + strideA * blockIdx.z;
  float* Cb = C + strideC * blockIdx.z;
  int m0 = blockIdx.y * 16;
  int n0 = blockIdx.x * (16 * NTILES);

  // TDM: LDS[0:16, 0:K] = A[m0:m0+16, 0:K]  (rows beyond M read as zero)
  tdm_load_2d_bf16((unsigned)(uintptr_t)&lds_a[0], Ab + (size_t)m0 * K,
                   K, 16, K, M - m0);
  __builtin_amdgcn_s_wait_tensorcnt((short)0);

  v8f acc[NTILES];
#pragma unroll
  for (int tn = 0; tn < NTILES; ++tn) { v8f z = {}; acc[tn] = z; }

  for (int k0 = 0; k0 < K; k0 += 32) {
    v16bf af = load_a_frag(lds_a, K, k0);
#pragma unroll
    for (int tn = 0; tn < NTILES; ++tn)
      acc[tn] = wmma_bf16(af, load_b_frag(W, K, n0 + 16 * tn, k0), acc[tn]);
  }

  int lane = threadIdx.x & 31;
  int mbase = m0 + ((lane >> 4) ? 8 : 0);
#pragma unroll
  for (int tn = 0; tn < NTILES; ++tn) {
    int n = n0 + 16 * tn + (lane & 15);
    float bv = bias ? bias[n] : 0.0f;
#pragma unroll
    for (int r = 0; r < 8; ++r) {
      int m = mbase + r;
      if (m < M) Cb[(size_t)m * ldc + n] = acc[tn][r] + bv;
    }
  }
}

// ------------------------------------------------- persistent LSTM+MHA scan
// One 1024-thread workgroup (32 wave32s). Batch dim M=8 padded to 16
// (rows 8..15 of the bf16 A tiles stay zero). State lives in LDS.
__global__ void __launch_bounds__(1024)
scan_kernel(const unsigned short* __restrict__ e_bf,
            const unsigned short* __restrict__ Wih,
            const unsigned short* __restrict__ Whh,
            const unsigned short* __restrict__ Wq,
            const float* __restrict__ b_ih,
            const float* __restrict__ b_hh,
            const float* __restrict__ ents,
            const int* __restrict__ entlens,
            const float* __restrict__ Kmat,
            const float* __restrict__ Vmat,
            float* __restrict__ l,
            unsigned short* __restrict__ l_bf) {
  __shared__ unsigned short prev_bf[16][H2_];   // 32 KB  [a | k]
  __shared__ unsigned short hx_bf[16][H_];      // 16 KB
  __shared__ float cxs[B_][H_];                 // 16 KB
  __shared__ float big[B_ * G4_];               // 64 KB gates; reused as q+scores
  __shared__ int elen[B_];

  int tid  = threadIdx.x;
  int lane = tid & 31;
  int wave = tid >> 5;

  for (int i = tid; i < 16 * H2_; i += 1024) (&prev_bf[0][0])[i] = 0;
  for (int i = tid; i < 16 * H_;  i += 1024) (&hx_bf[0][0])[i]  = 0;
  if (tid < B_) elen[tid] = entlens[tid];
  // hx0 = cx0 = mean over entities
  for (int i = tid; i < B_ * H_; i += 1024) {
    int b = i >> 9, h = i & (H_ - 1);
    float sum = 0.f;
    for (int n = 0; n < NE_; ++n) sum += ents[((size_t)b * NE_ + n) * H_ + h];
    float mean = sum * (1.f / NE_);
    cxs[b][h] = mean;
    hx_bf[b][h] = f32_to_bf16(mean);
  }
  __syncthreads();

  float* gates  = big;               // [B_][G4_]
  float* q      = big;               // [B_][H_]   (after gates consumed)
  float* scores = big + B_ * H_;     // [32][NE_]

  for (int t = 0; t < T_; ++t) {
    // Opaque memory clobber: keep the (L2-resident) weight-fragment loads
    // inside the timestep loop instead of LICM-hoisting them into a huge
    // register set that spills to scratch.
    asm volatile("" ::: "memory");

    // A: place k_t into prev[:, 512:1024]
    for (int i = tid; i < B_ * H_; i += 1024) {
      int b = i >> 9, h = i & (H_ - 1);
      prev_bf[b][H_ + h] = e_bf[((size_t)b * T_ + t) * H_ + h];
    }
    __syncthreads();

    // B: gates = prev @ W_ih^T + hx @ W_hh^T + biases  (128 N-tiles / 32 waves)
#pragma unroll
    for (int ti = 0; ti < 4; ++ti) {
      int n0 = (wave * 4 + ti) * 16;
      v8f acc = {};
#pragma clang loop unroll_count(4)
      for (int k0 = 0; k0 < H2_; k0 += 32)
        acc = wmma_bf16(load_a_frag(&prev_bf[0][0], H2_, k0),
                        load_b_frag(Wih, H2_, n0, k0), acc);
#pragma clang loop unroll_count(4)
      for (int k0 = 0; k0 < H_; k0 += 32)
        acc = wmma_bf16(load_a_frag(&hx_bf[0][0], H_, k0),
                        load_b_frag(Whh, H_, n0, k0), acc);
      int n = n0 + (lane & 15);
      float bias = b_ih[n] + b_hh[n];
      int mbase = (lane >> 4) ? 8 : 0;
#pragma unroll
      for (int r = 0; r < 8; ++r) {
        int m = mbase + r;
        if (m < B_) gates[m * G4_ + n] = acc[r] + bias;
      }
    }
    __syncthreads();

    // C: LSTM cell elementwise
    for (int i = tid; i < B_ * H_; i += 1024) {
      int b = i >> 9, h = i & (H_ - 1);
      float ig = gates[b * G4_ + h];
      float fg = gates[b * G4_ + H_ + h];
      float gg = gates[b * G4_ + 2 * H_ + h];
      float og = gates[b * G4_ + 3 * H_ + h];
      float c  = sigmoidf(fg) * cxs[b][h] + sigmoidf(ig) * tanhf(gg);
      float hx = sigmoidf(og) * tanhf(c);
      cxs[b][h]  = c;
      hx_bf[b][h] = f32_to_bf16(hx);
      size_t li = ((size_t)b * T_ + t) * H2_ + h;
      l[li]    = hx;
      l_bf[li] = f32_to_bf16(hx);
    }
    __syncthreads();

    // D: q = hx @ Wq^T   (32 N-tiles, one per wave)
    {
      int n0 = wave * 16;
      v8f acc = {};
#pragma clang loop unroll_count(4)
      for (int k0 = 0; k0 < H_; k0 += 32)
        acc = wmma_bf16(load_a_frag(&hx_bf[0][0], H_, k0),
                        load_b_frag(Wq, H_, n0, k0), acc);
      int n = n0 + (lane & 15);
      int mbase = (lane >> 4) ? 8 : 0;
#pragma unroll
      for (int r = 0; r < 8; ++r) {
        int m = mbase + r;
        if (m < B_) q[m * H_ + n] = acc[r];
      }
    }
    __syncthreads();

    // E: attention scores q . K  (+mask)
    for (int i = tid; i < B_ * NH_ * NE_; i += 1024) {
      int b = i >> 8, hh = (i >> 6) & 3, n = i & 63;
      const float* kp = Kmat + ((size_t)b * NE_ + n) * H_ + hh * HD_;
      const float* qp = q + b * H_ + hh * HD_;
      float sum = 0.f;
      for (int d = 0; d < HD_; ++d) sum += qp[d] * kp[d];
      sum *= 0.04419417382415922f;           // 1/sqrt(512)
      if (n > elen[b]) sum = -1e30f;
      scores[i] = sum;
    }
    __syncthreads();

    // F: softmax over NE + weighted V; one wave per (b, head)
    {
      int b = wave >> 2, hh = wave & 3;
      float* sr = scores + wave * NE_;
      float s0 = sr[lane], s1 = sr[lane + 32];
      float mx = fmaxf(s0, s1);
      for (int off = 16; off > 0; off >>= 1) mx = fmaxf(mx, __shfl_xor(mx, off, 32));
      float e0 = __expf(s0 - mx), e1 = __expf(s1 - mx);
      float sm = e0 + e1;
      for (int off = 16; off > 0; off >>= 1) sm += __shfl_xor(sm, off, 32);
      float inv = 1.f / sm;
      sr[lane] = e0 * inv; sr[lane + 32] = e1 * inv;
#pragma unroll
      for (int j = 0; j < 4; ++j) {
        int dd = lane + 32 * j;
        float av = 0.f;
        for (int n = 0; n < NE_; ++n)
          av += sr[n] * Vmat[((size_t)b * NE_ + n) * H_ + hh * HD_ + dd];
        int h = hh * HD_ + dd;
        size_t li = ((size_t)b * T_ + t) * H2_ + H_ + h;
        l[li]    = av;
        l_bf[li] = f32_to_bf16(av);
        prev_bf[b][h] = f32_to_bf16(av);     // a_t for next step
      }
    }
    __syncthreads();
  }
}

// ------------------------------------------------- switch gate s = sigmoid(l.sw_W)
__global__ void sw_kernel(const float* __restrict__ l, const float* __restrict__ swW,
                          const float* __restrict__ swb, float* __restrict__ s) {
  int w = (blockIdx.x * blockDim.x + threadIdx.x) >> 5;
  int lane = threadIdx.x & 31;
  if (w >= B_ * T_) return;
  const float* row = l + (size_t)w * H2_;
  float sum = 0.f;
  for (int k = lane; k < H2_; k += 32) sum += row[k] * swW[k];
  for (int off = 16; off > 0; off >>= 1) sum += __shfl_xor(sum, off, 32);
  if (lane == 0) s[w] = 1.f / (1.f + __expf(-(sum + swb[0])));
}

// ------------------------------------------------- pointer dist z over entities
__global__ void z_kernel(const float* __restrict__ dec, const float* __restrict__ ents,
                         const int* __restrict__ entlens, const float* __restrict__ s,
                         float* __restrict__ out0, float* __restrict__ zout) {
  __shared__ float red[NE_];
  int row = blockIdx.x, b = row / T_, n = threadIdx.x;
  const float* d = dec + (size_t)row * H_;
  const float* e = ents + ((size_t)b * NE_ + n) * H_;
  float u = 0.f;
  for (int h = 0; h < H_; ++h) u += d[h] * e[h];
  if (n > entlens[b]) u = -1e30f;
  red[n] = u; __syncthreads();
  for (int off = 32; off > 0; off >>= 1) { if (n < off) red[n] = fmaxf(red[n], red[n + off]); __syncthreads(); }
  float mx = red[0]; __syncthreads();
  float ev = __expf(u - mx);
  red[n] = ev; __syncthreads();
  for (int off = 32; off > 0; off >>= 1) { if (n < off) red[n] += red[n + off]; __syncthreads(); }
  float z = ev / red[0] * (1.f - s[row]);
  zout[(size_t)row * NE_ + n] = z;
  out0[(size_t)row * OUTC + V_ + n] = __logf(z + 1e-6f);
}

// ------------------------------------------------- in-place vocab softmax*s -> log
__global__ void vocab_softmax_kernel(float* __restrict__ out0, const float* __restrict__ s) {
  __shared__ float red[256];
  int row = blockIdx.x, tid = threadIdx.x;
  float* p = out0 + (size_t)row * OUTC;
  float mx = -1e30f;
  for (int i = tid; i < V_; i += 256) mx = fmaxf(mx, p[i]);
  red[tid] = mx; __syncthreads();
  for (int off = 128; off > 0; off >>= 1) { if (tid < off) red[tid] = fmaxf(red[tid], red[tid + off]); __syncthreads(); }
  mx = red[0]; __syncthreads();
  float sum = 0.f;
  for (int i = tid; i < V_; i += 256) sum += __expf(p[i] - mx);
  red[tid] = sum; __syncthreads();
  for (int off = 128; off > 0; off >>= 1) { if (tid < off) red[tid] += red[tid + off]; __syncthreads(); }
  float inv = 1.f / red[0];
  float sv = s[row];
  for (int i = tid; i < V_; i += 256) p[i] = __logf(__expf(p[i] - mx) * inv * sv + 1e-6f);
}

// ================================================================ launch
extern "C" void kernel_launch(void* const* d_in, const int* in_sizes, int n_in,
                              void* d_out, int out_size, void* d_ws, size_t ws_size,
                              hipStream_t stream) {
  (void)in_sizes; (void)n_in; (void)out_size; (void)ws_size;
  const int*   outp    = (const int*)  d_in[0];
  const float* ents    = (const float*)d_in[1];
  const int*   entlens = (const int*)  d_in[2];
  const float* emb     = (const float*)d_in[3];
  const float* W_ih    = (const float*)d_in[4];
  const float* W_hh    = (const float*)d_in[5];
  const float* b_ih    = (const float*)d_in[6];
  const float* b_hh    = (const float*)d_in[7];
  const float* Wq      = (const float*)d_in[8];
  const float* Wk      = (const float*)d_in[9];
  const float* Wv      = (const float*)d_in[10];
  const float* out_W   = (const float*)d_in[11];
  const float* out_b   = (const float*)d_in[12];
  const float* sw_W    = (const float*)d_in[13];
  const float* sw_b    = (const float*)d_in[14];
  const float* mattn_W = (const float*)d_in[15];
  const float* mattn_b = (const float*)d_in[16];

  char* ws = (char*)d_ws;
  size_t off = 0;
  auto alloc = [&](size_t bytes) {
    void* p = ws + off;
    off = (off + bytes + 255) & ~(size_t)255;
    return p;
  };
  unsigned short* e_bf      = (unsigned short*)alloc((size_t)B_ * T_ * H_ * 2);
  unsigned short* ents_bf   = (unsigned short*)alloc((size_t)B_ * NE_ * H_ * 2);
  unsigned short* Wih_bf    = (unsigned short*)alloc((size_t)G4_ * H2_ * 2);
  unsigned short* Whh_bf    = (unsigned short*)alloc((size_t)G4_ * H_ * 2);
  unsigned short* Wq_bf     = (unsigned short*)alloc((size_t)H_ * H_ * 2);
  unsigned short* Wk_bf     = (unsigned short*)alloc((size_t)H_ * H_ * 2);
  unsigned short* Wv_bf     = (unsigned short*)alloc((size_t)H_ * H_ * 2);
  unsigned short* outW_bf   = (unsigned short*)alloc((size_t)V_ * H2_ * 2);
  unsigned short* mattnW_bf = (unsigned short*)alloc((size_t)H_ * H2_ * 2);
  float* Kmat = (float*)alloc((size_t)B_ * NE_ * H_ * 4);
  float* Vmat = (float*)alloc((size_t)B_ * NE_ * H_ * 4);
  float* l    = (float*)alloc((size_t)B_ * T_ * H2_ * 4);
  unsigned short* l_bf = (unsigned short*)alloc((size_t)B_ * T_ * H2_ * 2);
  float* dec  = (float*)alloc((size_t)B_ * T_ * H_ * 4);
  float* sbuf = (float*)alloc((size_t)B_ * T_ * 4);

  float* out0 = (float*)d_out;
  float* zout = out0 + (size_t)B_ * T_ * OUTC;

  dim3 cg(2048), cb(256);
  to_bf16_kernel<<<cg, cb, 0, stream>>>(W_ih,    Wih_bf,    (long)G4_ * H2_);
  to_bf16_kernel<<<cg, cb, 0, stream>>>(W_hh,    Whh_bf,    (long)G4_ * H_);
  to_bf16_kernel<<<cg, cb, 0, stream>>>(Wq,      Wq_bf,     (long)H_ * H_);
  to_bf16_kernel<<<cg, cb, 0, stream>>>(Wk,      Wk_bf,     (long)H_ * H_);
  to_bf16_kernel<<<cg, cb, 0, stream>>>(Wv,      Wv_bf,     (long)H_ * H_);
  to_bf16_kernel<<<cg, cb, 0, stream>>>(out_W,   outW_bf,   (long)V_ * H2_);
  to_bf16_kernel<<<cg, cb, 0, stream>>>(mattn_W, mattnW_bf, (long)H_ * H2_);
  to_bf16_kernel<<<cg, cb, 0, stream>>>(ents,    ents_bf,   (long)B_ * NE_ * H_);

  gather_kernel<<<dim3(B_ * T_), dim3(256), 0, stream>>>(outp, emb, e_bf);

  // K = ents@Wk^T, V = ents@Wv^T   (batched over B; 4 N-tiles per wave)
  wmma_gemm_bt<4><<<dim3(H_ / 64, NE_ / 16, B_), dim3(32), 0, stream>>>(
      ents_bf, Wk_bf, nullptr, Kmat, NE_, H_, H_, H_, (long)NE_ * H_, (long)NE_ * H_);
  wmma_gemm_bt<4><<<dim3(H_ / 64, NE_ / 16, B_), dim3(32), 0, stream>>>(
      ents_bf, Wv_bf, nullptr, Vmat, NE_, H_, H_, H_, (long)NE_ * H_, (long)NE_ * H_);

  scan_kernel<<<dim3(1), dim3(1024), 0, stream>>>(
      e_bf, Wih_bf, Whh_bf, Wq_bf, b_ih, b_hh, ents, entlens, Kmat, Vmat, l, l_bf);

  sw_kernel<<<dim3(B_ * T_ / 8), dim3(256), 0, stream>>>(l, sw_W, sw_b, sbuf);

  // logits straight into d_out (row stride OUTC); TDM-staged A, 4 tiles/wave
  wmma_gemm_bt<4><<<dim3(V_ / 64, (B_ * T_) / 16, 1), dim3(32), 0, stream>>>(
      l_bf, outW_bf, out_b, out0, B_ * T_, V_, H2_, OUTC, 0, 0);

  // dec = l @ mattn_W^T + b
  wmma_gemm_bt<4><<<dim3(H_ / 64, (B_ * T_) / 16, 1), dim3(32), 0, stream>>>(
      l_bf, mattnW_bf, mattn_b, dec, B_ * T_, H_, H2_, H_, 0, 0);

  z_kernel<<<dim3(B_ * T_), dim3(NE_), 0, stream>>>(dec, ents, entlens, sbuf, out0, zout);

  vocab_softmax_kernel<<<dim3(B_ * T_), dim3(256), 0, stream>>>(out0, sbuf);
}
